// Attention__78786880078450
// MI455X (gfx1250) — matrix operand order repaired and checked by reading, not compiled, not run on hardware
//
#include <hip/hip_runtime.h>
#include <hip/hip_bf16.h>

// ---------------------------------------------------------------------------
// MI455X (gfx1250) Performer hybrid attention.
// All contractions run on V_WMMA_F32_16X16X32_BF16 (f32 accumulate), inputs
// converted fp32->bf16 inline while building lane fragments. Wave32 layouts
// per CDNA5 ISA 7.12.2:
//   A (16x32 bf16): lane l -> row (l&15); elem e   : K = (l>>4)*16+e (code's order; ISA table differs)
//   B (32x16 bf16): lane l -> col (l&15); elem e   : K = (l>>4)*16+e (code's order; ISA table differs)
//   C/D (16x16 f32): lane l -> col (l&15); vgpr v  : row = v + 8*(l>>4)
// Workspace requirement: ~270 MB fp32 (see kernel_launch layout).
// ---------------------------------------------------------------------------

typedef __attribute__((ext_vector_type(16))) __bf16 v16bf;
typedef __attribute__((ext_vector_type(8)))  float  v8f;

#define WMMA_BF16(A, Bf, C) \
  __builtin_amdgcn_wmma_f32_16x16x32_bf16(false, (A), false, (Bf), (short)0, (C), false, false)

static constexpr int  Bc  = 2;
static constexpr int  Nc  = 4096;
static constexpr int  Dc  = 1024;
static constexpr int  GHc = 8;
static constexpr int  DHc = 64;
static constexpr int  Mc  = 256;
static constexpr int  NWc = 16;   // windows per sequence (4096/256)

// Build A fragment from a contiguous fp32 row segment (two 8-elem runs).
__device__ inline v16bf a_frag(const float* p, float scale) {
  v16bf a;
#pragma unroll
  for (int e = 0; e < 8; ++e) a[e] = (__bf16)(p[e] * scale);
#pragma unroll
  for (int e = 0; e < 8; ++e) a[e + 8] = (__bf16)(p[8 + e] * scale);
  return a;
}

// ---------------------------------------------------------------------------
// Generic 8192x1024x1024 GEMM: C = A * W (+ bias).  Wave tile 32x64,
// workgroup (8 waves) tile 128x128.  grid = (64, 8), block = 256.
// ---------------------------------------------------------------------------
__global__ __launch_bounds__(256) void gemm_bf16(const float* __restrict__ A,
                                                 const float* __restrict__ Wt,
                                                 const float* __restrict__ bias,
                                                 float* __restrict__ C) {
  const int lane = threadIdx.x & 31, wave = threadIdx.x >> 5;
  const int half = lane >> 4, l15 = lane & 15;
  const int wm = wave & 3, wn = wave >> 2;
  const int row0 = blockIdx.x * 128 + wm * 32;
  const int col0 = blockIdx.y * 128 + wn * 64;

  v8f acc[2][4] = {};
  for (int k = 0; k < Dc; k += 32) {
    v16bf a0 = a_frag(A + (size_t)(row0 + l15) * Dc + k + half * 16, 1.0f);
    v16bf a1 = a_frag(A + (size_t)(row0 + 16 + l15) * Dc + k + half * 16, 1.0f);
#pragma unroll
    for (int j = 0; j < 4; ++j) {
      const float* bp = Wt + (size_t)(k + half * 16) * Dc + col0 + j * 16 + l15;
      v16bf bb;
#pragma unroll
      for (int e = 0; e < 16; ++e) bb[e] = (__bf16)bp[(size_t)e * Dc];
      acc[0][j] = WMMA_BF16(a0, bb, acc[0][j]);
      acc[1][j] = WMMA_BF16(a1, bb, acc[1][j]);
    }
  }
#pragma unroll
  for (int i = 0; i < 2; ++i)
#pragma unroll
    for (int j = 0; j < 4; ++j)
#pragma unroll
      for (int v = 0; v < 8; ++v) {
        const int r = row0 + i * 16 + v + 8 * half;
        const int c = col0 + j * 16 + l15;
        float val = acc[i][j][v];
        if (bias) val += bias[c];
        C[(size_t)r * Dc + c] = val;
      }
}

// ---------------------------------------------------------------------------
// data_dash[bh][n][m] = dn * sum_d data[b,n,h*64+d] * proj[m,d]
// grid = (N/128, M/64, B*GH), block = 256 (wave tile 16x64, K=64).
// ---------------------------------------------------------------------------
__global__ __launch_bounds__(256) void dash_kernel(const float* __restrict__ data,
                                                   const float* __restrict__ proj,
                                                   float* __restrict__ dash) {
  const int lane = threadIdx.x & 31, wave = threadIdx.x >> 5;
  const int half = lane >> 4, l15 = lane & 15;
  const int bh = blockIdx.z, b = bh >> 3, h = bh & 7;
  const int n0 = blockIdx.x * 128 + wave * 16;
  const int m0 = blockIdx.y * 64;
  const float dn = 0.35355339059327373f;  // 64^-0.25

  const float* arow = data + ((size_t)(b * Nc) + n0 + l15) * Dc + h * DHc;
  v8f acc[4] = {};
#pragma unroll
  for (int k = 0; k < 64; k += 32) {
    v16bf a = a_frag(arow + k + half * 16, dn);
#pragma unroll
    for (int j = 0; j < 4; ++j) {
      const float* pp = proj + (size_t)(m0 + j * 16 + l15) * DHc + k + half * 16;
      v16bf bb;
#pragma unroll
      for (int e = 0; e < 16; ++e) bb[e] = (__bf16)pp[e];
      acc[j] = WMMA_BF16(a, bb, acc[j]);
    }
  }
#pragma unroll
  for (int j = 0; j < 4; ++j)
#pragma unroll
    for (int v = 0; v < 8; ++v)
      dash[((size_t)bh * Nc + n0 + v + 8 * half) * Mc + m0 + j * 16 + l15] = acc[j][v];
}

// ---------------------------------------------------------------------------
// Deterministic 2-stage global max (for key stabilizer).
// ---------------------------------------------------------------------------
__global__ __launch_bounds__(256) void reduce_max1(const float* __restrict__ src,
                                                   long long n, float* __restrict__ partial) {
  float m = -3.0e38f;
  for (long long i = (long long)blockIdx.x * 256 + threadIdx.x; i < n;
       i += (long long)gridDim.x * 256)
    m = fmaxf(m, src[i]);
  __shared__ float sm[256];
  sm[threadIdx.x] = m;
  __syncthreads();
  for (int s = 128; s > 0; s >>= 1) {
    if ((int)threadIdx.x < s) sm[threadIdx.x] = fmaxf(sm[threadIdx.x], sm[threadIdx.x + s]);
    __syncthreads();
  }
  if (threadIdx.x == 0) partial[blockIdx.x] = sm[0];
}

__global__ __launch_bounds__(256) void reduce_max2(const float* __restrict__ partial,
                                                   int n, float* __restrict__ out) {
  float m = -3.0e38f;
  for (int i = threadIdx.x; i < n; i += 256) m = fmaxf(m, partial[i]);
  __shared__ float sm[256];
  sm[threadIdx.x] = m;
  __syncthreads();
  for (int s = 128; s > 0; s >>= 1) {
    if ((int)threadIdx.x < s) sm[threadIdx.x] = fmaxf(sm[threadIdx.x], sm[threadIdx.x + s]);
    __syncthreads();
  }
  if (threadIdx.x == 0) out[0] = sm[0];
}

// ---------------------------------------------------------------------------
// phi = ratio*(exp(dash - diag - stab) + eps), in place.  One wave per row
// (256 features); diag computed from the fp32 head row; q uses per-row max,
// k uses the precomputed global max.  grid = 8192, block = 256.
// ---------------------------------------------------------------------------
__global__ __launch_bounds__(256) void phi_exp_kernel(float* __restrict__ dash,
                                                      const float* __restrict__ data,
                                                      const float* __restrict__ kstab,
                                                      int is_query) {
  const int lane = threadIdx.x & 31, wave = threadIdx.x >> 5;
  const int row = blockIdx.x * 8 + wave;          // [0, B*GH*N)
  const int bh = row >> 12, n = row & (Nc - 1), b = bh >> 3, h = bh & 7;

  const float* drow = data + ((size_t)(b * Nc) + n) * Dc + h * DHc;
  float s = 0.f;
#pragma unroll
  for (int i = 0; i < 2; ++i) {
    float t = drow[lane + i * 32];
    s += t * t;
  }
  s += __shfl_xor(s, 1, 32);  s += __shfl_xor(s, 2, 32);
  s += __shfl_xor(s, 4, 32);  s += __shfl_xor(s, 8, 32);
  s += __shfl_xor(s, 16, 32);
  const float diag = s * 0.0625f;  // 0.5 * dn^2 = 1/16

  float* xr = dash + (size_t)row * Mc + lane * 8;
  float x[8];
#pragma unroll
  for (int i = 0; i < 8; ++i) x[i] = xr[i];

  float stab;
  if (is_query) {
    float m = x[0];
#pragma unroll
    for (int i = 1; i < 8; ++i) m = fmaxf(m, x[i]);
    m = fmaxf(m, __shfl_xor(m, 1, 32));  m = fmaxf(m, __shfl_xor(m, 2, 32));
    m = fmaxf(m, __shfl_xor(m, 4, 32));  m = fmaxf(m, __shfl_xor(m, 8, 32));
    m = fmaxf(m, __shfl_xor(m, 16, 32));
    stab = m;
  } else {
    stab = kstab[0];
  }
  const float ratio = 0.0625f;  // 256^-0.5
#pragma unroll
  for (int i = 0; i < 8; ++i) xr[i] = ratio * (__expf(x[i] - diag - stab) + 1.0e-4f);
}

// k_sum[bh][m] = sum_n phi_k[bh][n][m].  grid = 16, block = 256.
__global__ __launch_bounds__(256) void ksum_kernel(const float* __restrict__ phik,
                                                   float* __restrict__ ksum) {
  const int bh = blockIdx.x, m = threadIdx.x;
  const float* p = phik + (size_t)bh * Nc * Mc + m;
  float s = 0.f;
  for (int n = 0; n < Nc; ++n) s += p[(size_t)n * Mc];
  ksum[bh * Mc + m] = s;
}

// ---------------------------------------------------------------------------
// ctx[bh][m][d] = sum_n phi_k[bh][n][m] * V[b,n,h*64+d].  WMMA, K=N loop.
// grid = (2, 16), block = 256 (wave tile 16 m-rows x 64 d-cols).
// ---------------------------------------------------------------------------
__global__ __launch_bounds__(256) void ctx_kernel(const float* __restrict__ phik,
                                                  const float* __restrict__ V,
                                                  float* __restrict__ ctx) {
  const int lane = threadIdx.x & 31, wave = threadIdx.x >> 5;
  const int half = lane >> 4, l15 = lane & 15;
  const int bh = blockIdx.y, b = bh >> 3, h = bh & 7;
  const int m0 = blockIdx.x * 128 + wave * 16;

  const float* pk = phik + (size_t)bh * Nc * Mc;
  const float* vp = V + (size_t)(b * Nc) * Dc + h * DHc;
  v8f acc[4] = {};
  for (int n = 0; n < Nc; n += 32) {
    v16bf a;
    const int r = m0 + l15;
#pragma unroll
    for (int e = 0; e < 8; ++e) a[e] = (__bf16)pk[(size_t)(n + half * 16 + e) * Mc + r];
#pragma unroll
    for (int e = 0; e < 8; ++e) a[e + 8] = (__bf16)pk[(size_t)(n + 8 + half * 16 + e) * Mc + r];
#pragma unroll
    for (int j = 0; j < 4; ++j) {
      const float* q = vp + (size_t)(n + half * 16) * Dc + j * 16 + l15;
      v16bf bb;
#pragma unroll
      for (int e = 0; e < 16; ++e) bb[e] = (__bf16)q[(size_t)e * Dc];
      acc[j] = WMMA_BF16(a, bb, acc[j]);
    }
  }
#pragma unroll
  for (int j = 0; j < 4; ++j)
#pragma unroll
    for (int v = 0; v < 8; ++v)
      ctx[((size_t)bh * Mc + m0 + v + 8 * half) * DHc + j * 16 + l15] = acc[j][v];
}

// ---------------------------------------------------------------------------
// out_g[b,n,h*64+d] = (phi_q . ctx) / (phi_q . k_sum).  grid = (32,16).
// ---------------------------------------------------------------------------
__global__ __launch_bounds__(256) void gout_kernel(const float* __restrict__ phiq,
                                                   const float* __restrict__ ctx,
                                                   const float* __restrict__ ksum,
                                                   float* __restrict__ attn) {
  const int lane = threadIdx.x & 31, wave = threadIdx.x >> 5;
  const int half = lane >> 4, l15 = lane & 15;
  const int bh = blockIdx.y, b = bh >> 3, h = bh & 7;
  const int n0 = blockIdx.x * 128 + wave * 16;

  const float* pq = phiq + ((size_t)bh * Nc + n0) * Mc;

  // per-row normalizer: two lanes per row, 128 features each
  float sp = 0.f;
  {
    const float* rowp = pq + (size_t)l15 * Mc + half * 128;
    const float* kp = ksum + bh * Mc + half * 128;
    for (int i = 0; i < 128; ++i) sp += rowp[i] * kp[i];
  }
  sp += __shfl_xor(sp, 16, 32);
  __shared__ float srow[8][16];
  srow[wave][l15] = sp;  // both halves write identical value
  __syncthreads();

  v8f acc[4] = {};
  for (int m = 0; m < Mc; m += 32) {
    v16bf a = a_frag(pq + (size_t)l15 * Mc + m + half * 16, 1.0f);
#pragma unroll
    for (int j = 0; j < 4; ++j) {
      const float* cp = ctx + (size_t)bh * Mc * DHc + (size_t)(m + half * 16) * DHc + j * 16 + l15;
      v16bf bb;
#pragma unroll
      for (int e = 0; e < 16; ++e) bb[e] = (__bf16)cp[(size_t)e * DHc];
      acc[j] = WMMA_BF16(a, bb, acc[j]);
    }
  }
#pragma unroll
  for (int v = 0; v < 8; ++v) {
    const int r = v + 8 * half;
    const float dinv = 1.0f / srow[wave][r];
#pragma unroll
    for (int j = 0; j < 4; ++j)
      attn[((size_t)(b * Nc) + n0 + r) * Dc + h * DHc + j * 16 + l15] = acc[j][v] * dinv;
  }
}

// ---------------------------------------------------------------------------
// Local windowed attention, flash-style online softmax.  Each wave owns 16
// query rows; scores and PV both via WMMA; P staged through LDS (bf16) to
// re-layout C->A fragments, ordered with s_wait_dscnt.
// grid = (2, NW, B*LH), block = 256.
// ---------------------------------------------------------------------------
__global__ __launch_bounds__(256) void local_attn_kernel(const float* __restrict__ Qb,
                                                         const float* __restrict__ Kb,
                                                         const float* __restrict__ Vb,
                                                         float* __restrict__ attn) {
  const int lane = threadIdx.x & 31, wave = threadIdx.x >> 5;
  const int half = lane >> 4, l15 = lane & 15;
  const int bh = blockIdx.z, b = bh >> 3, hl = bh & 7, h = 8 + hl;
  const int iw = blockIdx.y;
  const int qn0 = iw * 256 + blockIdx.x * 128 + wave * 16;
  const float scale = 0.125f;  // 64^-0.5

  __shared__ __bf16 plds[8][16][32];

  v8f acc[4] = {};
  float mrow[8], lrow[8];
#pragma unroll
  for (int v = 0; v < 8; ++v) { mrow[v] = -1.0e30f; lrow[v] = 0.0f; }

  // q fragments (K = d = 64 -> two fragments), scale folded in
  const float* qrow = Qb + ((size_t)(b * Nc) + qn0 + l15) * Dc + h * DHc;
  v16bf qa[2];
#pragma unroll
  for (int t = 0; t < 2; ++t) qa[t] = a_frag(qrow + t * 32 + half * 16, scale);

  for (int jw = iw - 1; jw <= iw + 1; ++jw) {
    if (jw < 0 || jw >= NWc) continue;  // boundary windows: identical to -1e9 mask
    const int kwb = jw * 256;
    for (int kk = 0; kk < 256; kk += 32) {
      v8f s0 = {}, s1 = {};
#pragma unroll
      for (int t = 0; t < 2; ++t) {
        const float* kp0 = Kb + ((size_t)(b * Nc) + kwb + kk + l15) * Dc + h * DHc + t * 32 + half * 16;
        const float* kp1 = kp0 + (size_t)16 * Dc;
        v16bf b0, b1;
#pragma unroll
        for (int e = 0; e < 16; ++e) { b0[e] = (__bf16)kp0[e]; b1[e] = (__bf16)kp1[e]; }
        s0 = WMMA_BF16(qa[t], b0, s0);
        s1 = WMMA_BF16(qa[t], b1, s1);
      }
      // online softmax update (row r = v + 8*half; 16 lanes share a row group)
#pragma unroll
      for (int v = 0; v < 8; ++v) {
        float cm = fmaxf(s0[v], s1[v]);
        cm = fmaxf(cm, __shfl_xor(cm, 1, 32));
        cm = fmaxf(cm, __shfl_xor(cm, 2, 32));
        cm = fmaxf(cm, __shfl_xor(cm, 4, 32));
        cm = fmaxf(cm, __shfl_xor(cm, 8, 32));
        const float mn = fmaxf(mrow[v], cm);
        const float alpha = __expf(mrow[v] - mn);
        mrow[v] = mn;
        const float p0 = __expf(s0[v] - mn);
        const float p1 = __expf(s1[v] - mn);
        float rs = p0 + p1;
        rs += __shfl_xor(rs, 1, 32);
        rs += __shfl_xor(rs, 2, 32);
        rs += __shfl_xor(rs, 4, 32);
        rs += __shfl_xor(rs, 8, 32);
        lrow[v] = lrow[v] * alpha + rs;
#pragma unroll
        for (int j = 0; j < 4; ++j) acc[j][v] *= alpha;
        const int r = v + 8 * half;
        plds[wave][r][l15]      = (__bf16)p0;
        plds[wave][r][16 + l15] = (__bf16)p1;
      }
      // same-wave LDS ordering + compiler fence (CDNA5 split counter wait)
      asm volatile("s_wait_dscnt 0" ::: "memory");
      v16bf pa;
#pragma unroll
      for (int e = 0; e < 8; ++e) pa[e]     = plds[wave][l15][half * 16 + e];
#pragma unroll
      for (int e = 0; e < 8; ++e) pa[e + 8] = plds[wave][l15][8 + half * 16 + e];
      // PV over this 32-key chunk
#pragma unroll
      for (int j = 0; j < 4; ++j) {
        const float* vp = Vb + ((size_t)(b * Nc) + kwb + kk + half * 16) * Dc + h * DHc + j * 16 + l15;
        v16bf bv;
#pragma unroll
        for (int e = 0; e < 16; ++e) bv[e] = (__bf16)vp[(size_t)e * Dc];
        acc[j] = WMMA_BF16(pa, bv, acc[j]);
      }
    }
  }
#pragma unroll
  for (int v = 0; v < 8; ++v) {
    const int r = v + 8 * half;
    const float inv = 1.0f / lrow[v];
    const size_t base = ((size_t)(b * Nc) + qn0 + r) * Dc + h * DHc;
#pragma unroll
    for (int j = 0; j < 4; ++j) attn[base + j * 16 + l15] = acc[j][v] * inv;
  }
}

// ---------------------------------------------------------------------------
extern "C" void kernel_launch(void* const* d_in, const int* in_sizes, int n_in,
                              void* d_out, int out_size, void* d_ws, size_t ws_size,
                              hipStream_t stream) {
  (void)in_sizes; (void)n_in; (void)out_size; (void)ws_size;
  const float* x    = (const float*)d_in[0];
  const float* Wq   = (const float*)d_in[1];
  const float* Wk   = (const float*)d_in[2];
  const float* Wv   = (const float*)d_in[3];
  const float* Wo   = (const float*)d_in[4];
  const float* bo   = (const float*)d_in[5];
  const float* proj = (const float*)d_in[6];

  // Workspace layout (floats); total ~67.4M floats (~270 MB).
  float* ws   = (float*)d_ws;
  float* Qb   = ws;  ws += (size_t)Bc * Nc * Dc;        // 8388608
  float* Kb   = ws;  ws += (size_t)Bc * Nc * Dc;
  float* Vb   = ws;  ws += (size_t)Bc * Nc * Dc;
  float* phiq = ws;  ws += (size_t)Bc * GHc * Nc * Mc;  // 16777216
  float* phik = ws;  ws += (size_t)Bc * GHc * Nc * Mc;
  float* part = ws;  ws += 1024;
  float* kst  = ws;  ws += 1024;
  float* ksum = ws;  ws += (size_t)Bc * GHc * Mc;       // 4096
  float* ctx  = ws;  ws += (size_t)Bc * GHc * Mc * DHc; // 262144
  float* attn = ws;  ws += (size_t)Bc * Nc * Dc;

  const dim3 blk(256);

  // 1) QKV projections
  gemm_bf16<<<dim3(64, 8), blk, 0, stream>>>(x, Wq, nullptr, Qb);
  gemm_bf16<<<dim3(64, 8), blk, 0, stream>>>(x, Wk, nullptr, Kb);
  gemm_bf16<<<dim3(64, 8), blk, 0, stream>>>(x, Wv, nullptr, Vb);

  // 2) FAVOR+ feature logits
  dash_kernel<<<dim3(32, 4, 16), blk, 0, stream>>>(Qb, proj, phiq);
  dash_kernel<<<dim3(32, 4, 16), blk, 0, stream>>>(Kb, proj, phik);

  // 3) key stabilizer (global max) and exponentials
  reduce_max1<<<512, blk, 0, stream>>>(phik, (long long)Bc * GHc * Nc * Mc, part);
  reduce_max2<<<1, blk, 0, stream>>>(part, 512, kst);
  phi_exp_kernel<<<8192, blk, 0, stream>>>(phiq, Qb, kst, 1);
  phi_exp_kernel<<<8192, blk, 0, stream>>>(phik, Kb, kst, 0);

  // 4) linear attention
  ksum_kernel<<<16, blk, 0, stream>>>(phik, ksum);
  ctx_kernel<<<dim3(2, 16), blk, 0, stream>>>(phik, Vb, ctx);
  gout_kernel<<<dim3(32, 16), blk, 0, stream>>>(phiq, ctx, ksum, attn);

  // 5) local windowed attention (heads 8..15)
  local_attn_kernel<<<dim3(2, NWc, 16), blk, 0, stream>>>(Qb, Kb, Vb, attn);

  // 6) output projection + bias
  gemm_bf16<<<dim3(64, 8), blk, 0, stream>>>(attn, Wo, bo, (float*)d_out);
}
